// Model_56264071577991
// MI455X (gfx1250) — matrix operand order repaired and checked by reading, not compiled, not run on hardware
//
#include <hip/hip_runtime.h>
#include <hip/hip_bf16.h>
#include <stdint.h>

#define B_SZ 2048
#define T_SZ 29
#define I_SZ 2235
#define H_SZ 20
#define G_SZ 80              // 4*H
#define M_SZ (B_SZ * T_SZ)   // 59392, divisible by 64
#define KT   70              // ceil(2235/32); last chunk has 27 valid K

typedef __attribute__((ext_vector_type(16))) __bf16 v16bf;
typedef __attribute__((ext_vector_type(8)))  __bf16 v8bf;
typedef __attribute__((ext_vector_type(8)))  float  v8f;
typedef __attribute__((ext_vector_type(4)))  unsigned int v4u;
typedef __attribute__((ext_vector_type(8)))  int v8i;
typedef __attribute__((ext_vector_type(4)))  int v4i;

#define PITCH_A 36                   // dwords/row: 32 data + 4 TDM pad
#define A_TILE_DW (64 * PITCH_A)     // one A buffer, dwords

// ---------------------------------------------------------------------------
// Kernel 0: prepack W_ih -> bf16 in B-fragment order: Wb[kt][gate][k_local],
// K tail zero-filled. 70*80*32 = 179200 elements (358 KB, L2-resident).
// ---------------------------------------------------------------------------
__global__ __launch_bounds__(256) void pack_w(const float* __restrict__ W_ih,
                                              __bf16* __restrict__ Wb) {
  const int idx = blockIdx.x * 256 + threadIdx.x;
  if (idx >= KT * G_SZ * 32) return;
  const int k  = idx & 31;
  const int g  = (idx >> 5) % G_SZ;
  const int kt = idx / (G_SZ * 32);
  const int kk = kt * 32 + k;
  const float w = (kk < I_SZ) ? W_ih[g * I_SZ + kk] : 0.0f;
  Wb[idx] = (__bf16)w;
}

// TDM: 2D tile 32(K) x 64(M) fp32 -> LDS with 36-dword pitch (pad_interval=32dw,
// pad_amount=4dw). tensor_dim0 = remaining K so OOB tail reads return zero.
__device__ __forceinline__ void tdm_load_a(uint32_t lds_addr, uint64_t gaddr, int k0) {
  const uint32_t td0 = (uint32_t)(I_SZ - k0);
  v4u g0;
  g0[0] = 1u;                                       // count=1, user mode
  g0[1] = lds_addr;                                 // lds_addr
  g0[2] = (uint32_t)gaddr;                          // global_addr[31:0]
  g0[3] = (uint32_t)(gaddr >> 32) | (2u << 30);     // global_addr[56:32] | type=2
  v8i g1;
  g1[0] = (int)((2u << 16)      // data_size = 4B
              | (1u << 20)      // pad_enable
              | (4u << 22)      // pad_interval: 32 dwords
              | (3u << 25));    // pad_amount: 4 dwords
  g1[1] = (int)((td0 & 0xFFFFu) << 16);                   // tensor_dim0 lo
  g1[2] = (int)(((td0 >> 16) & 0xFFFFu) | (64u << 16));   // tensor_dim0 hi | tensor_dim1=64
  g1[3] = (int)(32u << 16);                               // tile_dim0 = 32
  g1[4] = 64;                                             // tile_dim1 = 64, tile_dim2 = 0
  g1[5] = I_SZ;                                           // tensor_dim0_stride
  g1[6] = 0;
  g1[7] = 0;
  v4i gz = (v4i){0, 0, 0, 0};
#if defined(__clang_major__) && (__clang_major__ >= 23)
  v8i gz8 = (v8i){0, 0, 0, 0, 0, 0, 0, 0};
  __builtin_amdgcn_tensor_load_to_lds(g0, g1, gz, gz, gz8, 0);
#else
  __builtin_amdgcn_tensor_load_to_lds(g0, g1, gz, gz, 0);
#endif
}

// ---------------------------------------------------------------------------
// Kernel 1: xg[m][g] = x[m][:] . W_ih[g][:] + (b_ih[g] + b_hh[g])
// 4 waves/block, each wave: one 16-row M tile x all 5 N tiles.
// Double-buffered TDM A staging overlapped with bf16 WMMA compute.
// ---------------------------------------------------------------------------
__global__ __launch_bounds__(128) void xg_gemm(const float* __restrict__ x,
                                               const __bf16* __restrict__ Wb,
                                               const float* __restrict__ b_ih,
                                               const float* __restrict__ b_hh,
                                               float* __restrict__ xg) {
  __shared__ __align__(16) float ldsA[2 * A_TILE_DW];

  const int tid  = threadIdx.x;
  const int lane = tid & 31;
  const int wave = tid >> 5;
  const int mBlock = blockIdx.x * 64;
  const int mWave  = mBlock + wave * 16;

  v8f acc[5];
#pragma unroll
  for (int t = 0; t < 5; ++t) acc[t] = (v8f){0.f,0.f,0.f,0.f,0.f,0.f,0.f,0.f};

  const uint32_t ldsA0 = (uint32_t)(uintptr_t)(&ldsA[0]);
  const uint64_t xBase = (uint64_t)(uintptr_t)x + (uint64_t)mBlock * (uint64_t)I_SZ * 4ull;

  // Per-lane invariants
  const int mrow = lane & 15;             // A: M row within tile
  const int h8   = (lane >> 4) * 8;       // A: K sub-offset selector
  const int ncol = lane & 15;             // B/D: N column within tile
  const int kg   = (lane >> 4) * 16;      // B: K half selector
  const float* arow = &ldsA[(wave * 16 + mrow) * PITCH_A];

  // Prologue: kick off tile 0 into buffer 0.
  if (wave == 0) tdm_load_a(ldsA0, xBase, 0);

  for (int kt = 0; kt < KT; ++kt) {
    // B fragments: two aligned 16B global loads per tile, straight from packed Wb.
    v16bf bfrag[5];
#pragma unroll
    for (int t = 0; t < 5; ++t) {
      const __bf16* wb = Wb + ((size_t)kt * G_SZ + t * 16 + ncol) * 32 + kg;
      v8bf lo = *(const v8bf*)(wb);
      v8bf hi = *(const v8bf*)(wb + 8);
      bfrag[t] = __builtin_shufflevector(lo, hi,
          0,1,2,3,4,5,6,7,8,9,10,11,12,13,14,15);
    }

    // Pipeline: issue TDM for kt+1 into the other buffer, then wait for tile kt.
    if (wave == 0) {
      if (kt + 1 < KT) {
        tdm_load_a(ldsA0 + (uint32_t)(((kt + 1) & 1) * A_TILE_DW * 4),
                   xBase + (uint64_t)(kt + 1) * 32ull * 4ull, (kt + 1) * 32);
        __builtin_amdgcn_s_wait_tensorcnt(1);   // in-order: tile kt has landed
      } else {
        __builtin_amdgcn_s_wait_tensorcnt(0);
      }
    }
    __syncthreads();

    // A fragment from LDS buffer (kt&1): lanes0-15 K{0-15}, lanes16-31 K{16-31}
    const float* ar = arow + (kt & 1) * A_TILE_DW;
    v16bf afrag;
#pragma unroll
    for (int i = 0; i < 8; ++i) {
      afrag[i]     = (__bf16)ar[2 * h8 + i];
      afrag[8 + i] = (__bf16)ar[2 * h8 + 8 + i];
    }

#pragma unroll
    for (int t = 0; t < 5; ++t)
      acc[t] = __builtin_amdgcn_wmma_f32_16x16x32_bf16(
          false, afrag, false, bfrag[t], (short)0, acc[t], false, false);

    __syncthreads();  // all waves done reading buf(kt&1) before TDM(kt+2) rewrites it
  }

  // Epilogue: + (b_ih + b_hh); C/D layout: VGPR r -> M = r + 8*(lane>=16), N = lane&15
  const int mhalf = (lane >> 4) * 8;
#pragma unroll
  for (int t = 0; t < 5; ++t) {
    const int g = t * 16 + ncol;
    const float bias = b_ih[g] + b_hh[g];
#pragma unroll
    for (int r = 0; r < 8; ++r) {
      const int row = mWave + mhalf + r;
      xg[(size_t)row * G_SZ + g] = acc[t][r] + bias;
    }
  }
}

// ---------------------------------------------------------------------------
// Kernel 2: LSTM scan + FC. Quad of lanes per batch element: lane q owns
// hidden j in [5q,5q+5) and its own i/f/g/o rows; h broadcast via __shfl.
// ---------------------------------------------------------------------------
__device__ __forceinline__ float sigmoid_f(float v) {
  return 1.0f / (1.0f + __expf(-v));
}
__device__ __forceinline__ float tanh_f(float v) {
  const float e = __expf(-2.0f * v);
  return (1.0f - e) / (1.0f + e);
}

__global__ __launch_bounds__(256) void lstm_scan(const float* __restrict__ xg,
                                                 const float* __restrict__ W_hh,
                                                 const float* __restrict__ W_fc,
                                                 const float* __restrict__ b_fc,
                                                 float* __restrict__ out) {
  __shared__ float wh[G_SZ * H_SZ];
  __shared__ float wfc[2 * H_SZ];
  __shared__ float bfc[2];
  for (int i = threadIdx.x; i < G_SZ * H_SZ; i += 256) wh[i] = W_hh[i];
  if (threadIdx.x < 2 * H_SZ) wfc[threadIdx.x] = W_fc[threadIdx.x];
  if (threadIdx.x < 2) bfc[threadIdx.x] = b_fc[threadIdx.x];
  __syncthreads();

  const int gid   = blockIdx.x * 256 + threadIdx.x;
  const int b     = gid >> 2;
  const int q     = gid & 3;
  const int lane  = threadIdx.x & 31;
  const int qbase = lane & ~3;

  float hall[H_SZ];       // full hidden state (replicated in quad)
  float cq[5];            // this lane's 5 cell states
#pragma unroll
  for (int j = 0; j < H_SZ; ++j) hall[j] = 0.f;
#pragma unroll
  for (int j = 0; j < 5; ++j) cq[j] = 0.f;

  const float* xrow = xg + (size_t)b * T_SZ * G_SZ;
  for (int t = 0; t < T_SZ; ++t) {
    const float* xt = xrow + t * G_SZ;
    float hq[5];
#pragma unroll
    for (int j = 0; j < 5; ++j) {
      const int jj = 5 * q + j;
      float gi = xt[jj];
      float gf = xt[H_SZ + jj];
      float gg = xt[2 * H_SZ + jj];
      float go = xt[3 * H_SZ + jj];
#pragma unroll
      for (int k = 0; k < H_SZ; ++k) {
        const float hv = hall[k];
        gi += hv * wh[jj * H_SZ + k];
        gf += hv * wh[(H_SZ + jj) * H_SZ + k];
        gg += hv * wh[(2 * H_SZ + jj) * H_SZ + k];
        go += hv * wh[(3 * H_SZ + jj) * H_SZ + k];
      }
      const float cn = sigmoid_f(gf) * cq[j] + sigmoid_f(gi) * tanh_f(gg);
      cq[j] = cn;
      hq[j] = sigmoid_f(go) * tanh_f(cn);
    }
    // Broadcast updated h across the quad (20 lane shuffles).
#pragma unroll
    for (int qq = 0; qq < 4; ++qq)
#pragma unroll
      for (int j = 0; j < 5; ++j)
        hall[qq * 5 + j] = __shfl(hq[j], qbase + qq, 32);
  }

  if (q == 0) {
    float o0 = bfc[0], o1 = bfc[1];
#pragma unroll
    for (int k = 0; k < H_SZ; ++k) {
      o0 += hall[k] * wfc[k];
      o1 += hall[k] * wfc[H_SZ + k];
    }
    out[(size_t)b * 2 + 0] = o0;
    out[(size_t)b * 2 + 1] = o1;
  }
}

extern "C" void kernel_launch(void* const* d_in, const int* in_sizes, int n_in,
                              void* d_out, int out_size, void* d_ws, size_t ws_size,
                              hipStream_t stream) {
  (void)in_sizes; (void)n_in; (void)out_size; (void)ws_size;
  const float* x    = (const float*)d_in[0];
  const float* W_ih = (const float*)d_in[1];
  const float* W_hh = (const float*)d_in[2];
  const float* b_ih = (const float*)d_in[3];
  const float* b_hh = (const float*)d_in[4];
  const float* W_fc = (const float*)d_in[5];
  const float* b_fc = (const float*)d_in[6];
  float* out = (float*)d_out;

  // Workspace layout: xg (59392*80 fp32 = 19,005,440 B) | Wb (70*80*32 bf16 = 358,400 B)
  float*  xg = (float*)d_ws;
  __bf16* Wb = (__bf16*)((char*)d_ws + (size_t)M_SZ * G_SZ * sizeof(float));

  pack_w<<<dim3((KT * G_SZ * 32 + 255) / 256), dim3(256), 0, stream>>>(W_ih, Wb);
  xg_gemm<<<dim3(M_SZ / 64), dim3(128), 0, stream>>>(x, Wb, b_ih, b_hh, xg);
  lstm_scan<<<dim3(B_SZ * 4 / 256), dim3(256), 0, stream>>>(xg, W_hh, W_fc, b_fc, out);
}
